// KBBlock_v5_25941602468455
// MI455X (gfx1250) — matrix operand order repaired and checked by reading, not compiled, not run on hardware
//
#include <hip/hip_runtime.h>
#include <math.h>

// ---------------------------------------------------------------------------
// KBBlock for MI455X (gfx1250): bf16 WMMA GEMMs for every 1x1 conv, fused
// affine/GELU staging + fused epilogues, LDS-tiled spatial convs,
// deterministic tree reductions for the 5 RMS norms.
// Round 3: epilogue options are compile-time template flags (no runtime
// branches in the hot loop) and epilogue I/O is vectorized (b128 loads/stores
// of the 8 consecutive pixels each lane owns).
// ---------------------------------------------------------------------------

typedef __bf16 bf16;
typedef __attribute__((ext_vector_type(16))) __bf16 v16bf;
typedef __attribute__((ext_vector_type(8)))  __bf16 v8bf;
typedef __attribute__((ext_vector_type(8)))  float  v8f;
typedef __attribute__((ext_vector_type(4)))  float  v4f;

#define HW 65536
#define IMW 256
#define NSAMP 8

static __device__ __forceinline__ float b2f(bf16 x) { return (float)x; }
static __device__ __forceinline__ bf16  f2b(float x) { return (bf16)x; }
static __device__ __forceinline__ float gelu_exact(float x) {
  return 0.5f * x * (1.0f + erff(x * 0.70710678118654752440f));
}
static __device__ __forceinline__ v16bf cat8(v8bf lo, v8bf hi) {
  return __builtin_shufflevector(lo, hi, 0, 1, 2, 3, 4, 5, 6, 7,
                                 8, 9, 10, 11, 12, 13, 14, 15);
}

// ---------------------------------------------------------------------------
// Generic 1x1-conv GEMM:  out[b, co, p] = epilogue( sum_ci A[b, ci, p] * W[co, ci] )
// A staged in LDS as [pixel][K] bf16 (optional f32->affine->gelu on the fly).
// WMMA 16x16x32 bf16, f32 accumulate. Block = 256 thr = 8 waves = 128 pixels.
// Fragment layouts per CDNA5 ISA 7.12.2:
//   A 16x32 bf16 : lanes 0-15 row M=lane, K{0..7,16..23}; lanes 16-31 K{8..15,24..31}
//   B 32x16 bf16 : lanes 0-15 col N=lane, K{0..7,16..23}; lanes 16-31 K{8..15,24..31}
//   D 16x16 f32  : lanes 0-15 N=lane, M=v; lanes 16-31 N=lane-16, M=8+v
// All knobs are compile-time so the hot loop is branch-free straight-line code.
// ---------------------------------------------------------------------------
template <int CINPAD, bool IN_F32, bool IN_GELU, bool OUT_F32,
          bool HAS_AFF, bool HAS_ES, bool HAS_RADD, bool HAS_RESF>
__global__ __launch_bounds__(256) void gemm1x1_kernel(
    const void* __restrict__ inPtr, int cinReal, long inSS,
    const float* __restrict__ inw, const float* __restrict__ inb,
    const float* __restrict__ inScale,               // per-sample scalar (rms)
    const bf16* __restrict__ W,                      // [cout][CINPAD] bf16
    const float* __restrict__ bias, int cout,
    void* __restrict__ outPtr, long outSS,
    const float* __restrict__ es,                    // per-co scale
    const bf16* __restrict__ radd, long raddSS,      // bf16 residual
    const float* __restrict__ resf, long resSS)      // f32 residual
{
  constexpr int NK = CINPAD / 32;
  __shared__ __align__(16) bf16 ldsA[128 * CINPAD];

  const int b    = blockIdx.x >> 9;                  // 512 blocks / sample
  const int pix0 = (blockIdx.x & 511) << 7;          // 128 pixels / block
  const int tid  = threadIdx.x;

  // ---- stage A tile: 128 pixels x CINPAD (bf16, K contiguous) ----
  const float sInScale = HAS_AFF ? inScale[b] : 1.0f;
#pragma unroll 4
  for (int e = tid; e < (CINPAD << 7); e += 256) {
    const int pix = e & 127;
    const int ci  = e >> 7;
    float v = 0.0f;
    if (ci < cinReal) {
      if (IN_F32)
        v = ((const float*)inPtr)[(long)b * inSS + (long)ci * HW + pix0 + pix];
      else
        v = b2f(((const bf16*)inPtr)[(long)b * inSS + (long)ci * HW + pix0 + pix]);
      if (HAS_AFF) v = v * (inw[ci] * sInScale) + inb[ci];
      if (IN_GELU) v = gelu_exact(v);
    }
    ldsA[pix * CINPAD + ci] = f2b(v);
  }
  __syncthreads();

  const int lane = tid & 31;
  const int wave = tid >> 5;
  const int row  = lane & 15;
  const int hi   = lane >> 4;
  const int arow = (wave << 4) + row;

  // ---- A fragments for all K tiles (static indices) ----
  v16bf afrag[NK];
#pragma unroll
  for (int kt = 0; kt < NK; ++kt) {
    const int k0 = (kt << 5) + (hi << 3);
    v8bf lo  = *(const v8bf*)&ldsA[arow * CINPAD + k0];
    v8bf hi8 = *(const v8bf*)&ldsA[arow * CINPAD + k0 + 16];
    afrag[kt] = cat8(lo, hi8);
  }

  // ---- N-tile loop over output channels ----
  const int pm = pix0 + (wave << 4) + (hi << 3);
  for (int nt = 0; nt < (cout >> 4); ++nt) {
    const int co = (nt << 4) + row;                  // D layout: N = lane&15
    v8f acc = {};
    const bf16* wrow = W + (long)co * CINPAD;
#pragma unroll
    for (int kt = 0; kt < NK; ++kt) {
      const int kb = (kt << 5) + (hi << 3);
      v8bf b0 = *(const v8bf*)&wrow[kb];
      v8bf b1 = *(const v8bf*)&wrow[kb + 16];
      acc = __builtin_amdgcn_wmma_f32_16x16x32_bf16(
          false, afrag[kt], false, cat8(b0, b1), (short)0, acc, false, false);
    }
    const float bia = bias[co];
    const float esv = HAS_ES ? es[co] : 1.0f;
    const long pbase = (long)co * HW + pm;           // 8 consecutive pixels / lane

    v8bf rv;                                         // vector epilogue operands
    if (HAS_RADD) rv = *(const v8bf*)&radd[(long)b * raddSS + pbase];
    v4f f0, f1;
    if (HAS_RESF) {
      f0 = *(const v4f*)&resf[(long)b * resSS + pbase];
      f1 = *(const v4f*)&resf[(long)b * resSS + pbase + 4];
    }

    float t[8];
#pragma unroll
    for (int v = 0; v < 8; ++v) {
      float x = acc[v] + bia;
      if (HAS_ES)   x *= esv;
      if (HAS_RADD) x += b2f(rv[v]);
      if (HAS_RESF) x += (v < 4) ? f0[v] : f1[v - 4];
      t[v] = x;
    }
    if (OUT_F32) {
      float* op = (float*)outPtr + (long)b * outSS + pbase;
      v4f o0 = {t[0], t[1], t[2], t[3]};
      v4f o1 = {t[4], t[5], t[6], t[7]};
      *(v4f*)op       = o0;
      *(v4f*)(op + 4) = o1;
    } else {
      v8bf ov;
#pragma unroll
      for (int v = 0; v < 8; ++v) ov[v] = f2b(t[v]);
      *(v8bf*)((bf16*)outPtr + (long)b * outSS + pbase) = ov;
    }
  }
}

// ---------------------------------------------------------------------------
// Grouped 5x5 conv (groups=16, 4 in-ch / 4 out-ch per group), pad=2, f32 out.
// One block = one (sample, group, 16x16 tile).
// ---------------------------------------------------------------------------
__global__ __launch_bounds__(256) void grouped5x5_kernel(
    const bf16* __restrict__ in, long inSS,
    const float* __restrict__ w,        // (64,4,5,5)
    const float* __restrict__ bias,
    float* __restrict__ out, long outSS)
{
  __shared__ bf16 tile[4 * 20 * 20];
  const int b = blockIdx.z, g = blockIdx.y;
  const int tX = (blockIdx.x & 15) << 4, tY = (blockIdx.x >> 4) << 4;
  const int tid = threadIdx.x, tx = tid & 15, ty = tid >> 4;

  for (int idx = tid; idx < 4 * 400; idx += 256) {
    const int c = idx / 400, rem = idx % 400, r = rem / 20, cc = rem % 20;
    const int gy = tY + r - 2, gx = tX + cc - 2;
    bf16 v = f2b(0.0f);
    if ((unsigned)gy < 256u && (unsigned)gx < 256u)
      v = in[(long)b * inSS + (long)(g * 4 + c) * HW + gy * IMW + gx];
    tile[idx] = v;
  }
  __syncthreads();

#pragma unroll
  for (int oc = 0; oc < 4; ++oc) {
    const int c = g * 4 + oc;
    float acc = bias[c];
    const float* wc = w + c * 100;
#pragma unroll
    for (int ic = 0; ic < 4; ++ic)
#pragma unroll
      for (int ky = 0; ky < 5; ++ky)
#pragma unroll
        for (int kx = 0; kx < 5; ++kx)
          acc += b2f(tile[ic * 400 + (ty + ky) * 20 + (tx + kx)]) *
                 wc[ic * 25 + ky * 5 + kx];
    out[(long)b * outSS + (long)c * HW + (tY + ty) * IMW + tX + tx] = acc;
  }
}

// ---------------------------------------------------------------------------
// kba2 grouped 5x5 + KBA combine:  xf = (k*ga1 + uf) * x1 * sca  (bf16 out)
// ---------------------------------------------------------------------------
__global__ __launch_bounds__(256) void kba2_combine_kernel(
    const bf16* __restrict__ k1, long k1SS,
    const float* __restrict__ w, const float* __restrict__ bias,
    const float* __restrict__ ga1,
    const float* __restrict__ uf, const float* __restrict__ x1,
    const float* __restrict__ sca,
    bf16* __restrict__ xf)
{
  __shared__ bf16 tile[4 * 20 * 20];
  const int b = blockIdx.z, g = blockIdx.y;
  const int tX = (blockIdx.x & 15) << 4, tY = (blockIdx.x >> 4) << 4;
  const int tid = threadIdx.x, tx = tid & 15, ty = tid >> 4;

  for (int idx = tid; idx < 4 * 400; idx += 256) {
    const int c = idx / 400, rem = idx % 400, r = rem / 20, cc = rem % 20;
    const int gy = tY + r - 2, gx = tX + cc - 2;
    bf16 v = f2b(0.0f);
    if ((unsigned)gy < 256u && (unsigned)gx < 256u)
      v = k1[(long)b * k1SS + (long)(g * 4 + c) * HW + gy * IMW + gx];
    tile[idx] = v;
  }
  __syncthreads();

#pragma unroll
  for (int oc = 0; oc < 4; ++oc) {
    const int c = g * 4 + oc;
    float acc = bias[c];
    const float* wc = w + c * 100;
#pragma unroll
    for (int ic = 0; ic < 4; ++ic)
#pragma unroll
      for (int ky = 0; ky < 5; ++ky)
#pragma unroll
        for (int kx = 0; kx < 5; ++kx)
          acc += b2f(tile[ic * 400 + (ty + ky) * 20 + (tx + kx)]) *
                 wc[ic * 25 + ky * 5 + kx];
    const long idx = (long)b * 64 * HW + (long)c * HW + (tY + ty) * IMW + tX + tx;
    const float v = (acc * ga1[c] + uf[idx]) * x1[idx] * sca[b * 64 + c];
    xf[idx] = f2b(v);
  }
}

// ---------------------------------------------------------------------------
// Depthwise 3x3, pad=1. Optional input norm+GELU (fdw path). Dual outputs.
// ---------------------------------------------------------------------------
template <bool NG>
__global__ __launch_bounds__(256) void dw3x3_kernel(
    const bf16* __restrict__ in, long inSS,
    const float* __restrict__ aw, const float* __restrict__ ab,
    const float* __restrict__ ascale,
    const float* __restrict__ w, const float* __restrict__ bias,
    float* __restrict__ outF, long outFSS,
    bf16* __restrict__ outB, long outBSS)
{
  __shared__ float tile[18 * 18];
  const int b = blockIdx.z, c = blockIdx.y;
  const int tX = (blockIdx.x & 15) << 4, tY = (blockIdx.x >> 4) << 4;
  const int tid = threadIdx.x, tx = tid & 15, ty = tid >> 4;
  const float s = NG ? ascale[b] : 0.0f;

  for (int idx = tid; idx < 324; idx += 256) {
    const int r = idx / 18, cc = idx % 18;
    const int gy = tY + r - 1, gx = tX + cc - 1;
    float v = 0.0f;
    if ((unsigned)gy < 256u && (unsigned)gx < 256u) {
      v = b2f(in[(long)b * inSS + (long)c * HW + gy * IMW + gx]);
      if (NG) { v = aw[c] * v * s + ab[c]; v = gelu_exact(v); }
    }
    tile[idx] = v;
  }
  __syncthreads();

  float acc = bias[c];
  const float* wc = w + c * 9;
#pragma unroll
  for (int ky = 0; ky < 3; ++ky)
#pragma unroll
    for (int kx = 0; kx < 3; ++kx)
      acc += tile[(ty + ky) * 18 + (tx + kx)] * wc[ky * 3 + kx];

  const long o = (long)c * HW + (tY + ty) * IMW + tX + tx;
  if (outF) outF[(long)b * outFSS + o] = acc;
  if (outB) outB[(long)b * outBSS + o] = f2b(acc);
}

// ---------------------------------------------------------------------------
// c2a grouped 3x3 (groups=32, 2in->1out) on rms-normed input + SimpleGate.
// sg[j] = t[j] * t[j+16]   (16 channels, bf16 out)
// ---------------------------------------------------------------------------
__global__ __launch_bounds__(256) void gate_kernel(
    const float* __restrict__ inp,
    const float* __restrict__ n1w, const float* __restrict__ n1b,
    const float* __restrict__ rms1,
    const float* __restrict__ w,      // (32,2,3,3)
    const float* __restrict__ bias,
    bf16* __restrict__ sg)
{
  __shared__ bf16 xt[64 * 18 * 18];   // 41.5 KB
  const int b = blockIdx.z;
  const int tX = (blockIdx.x & 15) << 4, tY = (blockIdx.x >> 4) << 4;
  const int tid = threadIdx.x, tx = tid & 15, ty = tid >> 4;
  const float rs = rms1[b];

  for (int idx = tid; idx < 64 * 324; idx += 256) {
    const int c = idx / 324, rem = idx % 324, r = rem / 18, cc = rem % 18;
    const int gy = tY + r - 1, gx = tX + cc - 1;
    float v = 0.0f;
    if ((unsigned)gy < 256u && (unsigned)gx < 256u)
      v = n1w[c] * inp[(long)b * 64 * HW + (long)c * HW + gy * IMW + gx] * rs + n1b[c];
    xt[idx] = f2b(v);
  }
  __syncthreads();

  float t[32];
#pragma unroll
  for (int o = 0; o < 32; ++o) {
    float acc = bias[o];
#pragma unroll
    for (int i = 0; i < 2; ++i) {
      const int base = (2 * o + i) * 324;
      const float* wc = w + (o * 2 + i) * 9;
#pragma unroll
      for (int ky = 0; ky < 3; ++ky)
#pragma unroll
        for (int kx = 0; kx < 3; ++kx)
          acc += b2f(xt[base + (ty + ky) * 18 + (tx + kx)]) * wc[ky * 3 + kx];
    }
    t[o] = acc;
  }
  const long pos = (tY + ty) * IMW + tX + tx;
#pragma unroll
  for (int j = 0; j < 16; ++j)
    sg[(long)b * 16 * HW + (long)j * HW + pos] = f2b(t[j] * t[j + 16]);
}

// ---------------------------------------------------------------------------
// Deterministic per-(b,c) reduction: sum of squares (and optional sum).
// One block per (b,c); fixed-order strided accumulation + LDS tree.
// ---------------------------------------------------------------------------
template <typename T>
__global__ __launch_bounds__(256) void sumsq_kernel(
    const T* __restrict__ in, long sampSS, int C,
    float* __restrict__ ssq, float* __restrict__ sum)
{
  const int b = blockIdx.x / C, c = blockIdx.x % C;
  const T* p = in + (long)b * sampSS + (long)c * HW;
  float a = 0.0f, s = 0.0f;
  for (int i = threadIdx.x; i < HW; i += 256) {
    const float v = (float)p[i];
    a += v * v; s += v;
  }
  __shared__ float l1[256], l2[256];
  l1[threadIdx.x] = a; l2[threadIdx.x] = s;
  __syncthreads();
  for (int off = 128; off; off >>= 1) {
    if (threadIdx.x < off) {
      l1[threadIdx.x] += l1[threadIdx.x + off];
      l2[threadIdx.x] += l2[threadIdx.x + off];
    }
    __syncthreads();
  }
  if (threadIdx.x == 0) {
    ssq[blockIdx.x] = l1[0];
    if (sum) sum[blockIdx.x] = l2[0];
  }
}

// n1 rms + SCA gate (per-channel means pass through the linear norm).
__global__ void finalize1_kernel(
    const float* __restrict__ ssq, const float* __restrict__ sum,
    const float* __restrict__ n1w, const float* __restrict__ n1b,
    const float* __restrict__ scaw, const float* __restrict__ scab,
    float* __restrict__ rms1, float* __restrict__ sca)
{
  const int b = blockIdx.x, c = threadIdx.x;   // 64 threads
  __shared__ float xm[64];
  float tot = 0.0f;
  for (int i = 0; i < 64; ++i) tot += ssq[b * 64 + i];
  const float r = rsqrtf(tot / (64.0f * (float)HW) + 1e-6f);
  if (c == 0) rms1[b] = r;
  xm[c] = n1w[c] * (sum[b * 64 + c] * (1.0f / (float)HW)) * r + n1b[c];
  __syncthreads();
  float g = scab[c];
  for (int i = 0; i < 64; ++i) g += scaw[c * 64 + i] * xm[i];
  sca[b * 64 + c] = g;
}

__global__ void finalize_rms_kernel(const float* __restrict__ ssq, int C,
                                    float* __restrict__ out)
{
  const int b = blockIdx.x;
  if (threadIdx.x == 0) {
    float t = 0.0f;
    for (int i = 0; i < C; ++i) t += ssq[b * C + i];
    out[b] = rsqrtf(t / ((float)C * (float)HW) + 1e-6f);
  }
}

// ---------------------------------------------------------------------------
// Pack GEMM weight matrices into bf16 [Cout][Cin] (W1 = concat c11a|c1|c211;
// c2c zero-padded K 16->32).
// ---------------------------------------------------------------------------
__global__ void pack_weights_kernel(
    const float* c11a, const float* c11ab, const float* c1w, const float* c1b,
    const float* c211, const float* c211b, const float* c2c,
    const float* kba1, const float* c3, const float* fe, const float* fp,
    bf16* W1, float* b1, bf16* Wc2c, bf16* Wkba1, bf16* Wc3, bf16* Wfe, bf16* Wfp)
{
  const int tid = threadIdx.x;
  for (int i = tid; i < 64 * 64; i += 256) { W1[i] = f2b(c11a[i]); W1[4096 + i] = f2b(c1w[i]); }
  for (int i = tid; i < 32 * 64; i += 256) W1[8192 + i] = f2b(c211[i]);
  for (int i = tid; i < 160; i += 256)
    b1[i] = (i < 64) ? c11ab[i] : ((i < 128) ? c1b[i - 64] : c211b[i - 128]);
  for (int i = tid; i < 32 * 32; i += 256) {
    const int co = i >> 5, k = i & 31;
    Wc2c[i] = (k < 16) ? f2b(c2c[co * 16 + k]) : f2b(0.0f);
  }
  for (int i = tid; i < 64 * 96; i += 256) Wkba1[i] = f2b(kba1[i]);
  for (int i = tid; i < 64 * 64; i += 256) Wc3[i] = f2b(c3[i]);
  for (int i = tid; i < 128 * 64; i += 256) Wfe[i] = f2b(fe[i]);
  for (int i = tid; i < 64 * 128; i += 256) Wfp[i] = f2b(fp[i]);
}

// out = y + (norm2(y) + norm3(p)*fls) * gamma
__global__ __launch_bounds__(256) void final_kernel(
    const float* __restrict__ y, const bf16* __restrict__ p,
    const float* __restrict__ n2w, const float* __restrict__ n2b,
    const float* __restrict__ rms2,
    const float* __restrict__ fn3w, const float* __restrict__ fn3b,
    const float* __restrict__ rmsP,
    const float* __restrict__ fls, const float* __restrict__ gamma,
    float* __restrict__ out)
{
  const long i = (long)blockIdx.x * 256 + threadIdx.x;
  const int c = (int)((i >> 16) & 63);
  const int b = (int)(i >> 22);
  const float yv = y[i];
  const float xn2 = n2w[c] * yv * rms2[b] + n2b[c];
  const float pv = b2f(p[i]);
  const float h3 = (fn3w[c] * pv * rmsP[b] + fn3b[c]) * fls[c];
  out[i] = yv + (xn2 + h3) * gamma[c];
}

// ---------------------------------------------------------------------------
extern "C" void kernel_launch(void* const* d_in, const int* in_sizes, int n_in,
                              void* d_out, int out_size, void* d_ws, size_t ws_size,
                              hipStream_t stream)
{
  (void)in_sizes; (void)n_in; (void)out_size; (void)ws_size;
  const float* inp    = (const float*)d_in[0];
  const float* n1w    = (const float*)d_in[1];
  const float* n1b    = (const float*)d_in[2];
  const float* n2w    = (const float*)d_in[3];
  const float* n2b    = (const float*)d_in[4];
  const float* scaw   = (const float*)d_in[5];
  const float* scab   = (const float*)d_in[6];
  const float* c11aw  = (const float*)d_in[7];
  const float* c11abv = (const float*)d_in[8];
  const float* c11bw  = (const float*)d_in[9];
  const float* c11bb  = (const float*)d_in[10];
  const float* c1w    = (const float*)d_in[11];
  const float* c1b    = (const float*)d_in[12];
  const float* c21w   = (const float*)d_in[13];
  const float* c21b   = (const float*)d_in[14];
  const float* kba1w  = (const float*)d_in[15];
  const float* kba1b  = (const float*)d_in[16];
  const float* kba2w  = (const float*)d_in[17];
  const float* kba2b  = (const float*)d_in[18];
  const float* c2aw   = (const float*)d_in[19];
  const float* c2ab   = (const float*)d_in[20];
  const float* c2cw   = (const float*)d_in[21];
  const float* c2cb   = (const float*)d_in[22];
  const float* c211w  = (const float*)d_in[23];
  const float* c211b  = (const float*)d_in[24];
  const float* c3w    = (const float*)d_in[25];
  const float* c3b    = (const float*)d_in[26];
  const float* few    = (const float*)d_in[27];
  const float* feb    = (const float*)d_in[28];
  const float* fn1w   = (const float*)d_in[29];
  const float* fn1b   = (const float*)d_in[30];
  const float* fdww   = (const float*)d_in[31];
  const float* fdwb   = (const float*)d_in[32];
  const float* fn2w   = (const float*)d_in[33];
  const float* fn2b   = (const float*)d_in[34];
  const float* fpw    = (const float*)d_in[35];
  const float* fpb    = (const float*)d_in[36];
  const float* fn3w   = (const float*)d_in[37];
  const float* fn3b   = (const float*)d_in[38];
  const float* fls    = (const float*)d_in[39];
  const float* ga1    = (const float*)d_in[40];
  const float* attg   = (const float*)d_in[41];
  const float* beta   = (const float*)d_in[42];
  const float* gamma  = (const float*)d_in[43];

  char* ws = (char*)d_ws;
  size_t off = 0;
  auto alloc = [&](size_t bytes) -> size_t {
    size_t o = off;
    off = (off + bytes + 255) & ~(size_t)255;
    return o;
  };
  const long P = (long)NSAMP * HW;

  const size_t o_rsum = alloc(1024 * 4);
  const size_t o_rssq = alloc(1024 * 4);
  const size_t o_rms1 = alloc(8 * 4);
  const size_t o_sca  = alloc(512 * 4);
  const size_t o_rms2 = alloc(8 * 4);
  const size_t o_rmsE = alloc(8 * 4);
  const size_t o_rmsD = alloc(8 * 4);
  const size_t o_rmsP = alloc(8 * 4);
  const size_t o_W1   = alloc(160 * 64 * 2);
  const size_t o_b1   = alloc(160 * 4);
  const size_t o_Wc2c = alloc(32 * 32 * 2);
  const size_t o_Wkb1 = alloc(64 * 96 * 2);
  const size_t o_Wc3  = alloc(64 * 64 * 2);
  const size_t o_Wfe  = alloc(128 * 64 * 2);
  const size_t o_Wfp  = alloc(64 * 128 * 2);
  const size_t o_g1   = alloc((size_t)P * 160 * 2);  // t1|t2|att0, bf16
  const size_t o_sg   = alloc((size_t)P * 16 * 2);
  const size_t o_uf   = alloc((size_t)P * 64 * 4);
  const size_t o_x1   = alloc((size_t)P * 64 * 4);
  const size_t o_cat  = alloc((size_t)P * 96 * 2);   // att(0..31)|uf(32..95)
  const size_t o_k1   = alloc((size_t)P * 64 * 2);
  const size_t o_xf   = alloc((size_t)P * 64 * 2);
  const size_t o_y    = alloc((size_t)P * 64 * 4);
  // aliases after the attention half is dead:
  const size_t o_e = o_g1;   // 128ch bf16 fits in g1 (160ch bf16)
  const size_t o_d = o_uf;   // 128ch bf16 == 64ch f32 bytes
  const size_t o_p = o_k1;

  float* rsum = (float*)(ws + o_rsum);
  float* rssq = (float*)(ws + o_rssq);
  float* rms1 = (float*)(ws + o_rms1);
  float* sca  = (float*)(ws + o_sca);
  float* rms2 = (float*)(ws + o_rms2);
  float* rmsE = (float*)(ws + o_rmsE);
  float* rmsD = (float*)(ws + o_rmsD);
  float* rmsP = (float*)(ws + o_rmsP);
  bf16* W1   = (bf16*)(ws + o_W1);
  float* b1  = (float*)(ws + o_b1);
  bf16* Wc2c = (bf16*)(ws + o_Wc2c);
  bf16* Wkb1 = (bf16*)(ws + o_Wkb1);
  bf16* Wc3  = (bf16*)(ws + o_Wc3);
  bf16* Wfe  = (bf16*)(ws + o_Wfe);
  bf16* Wfp  = (bf16*)(ws + o_Wfp);
  bf16* g1   = (bf16*)(ws + o_g1);
  bf16* sg   = (bf16*)(ws + o_sg);
  float* uf  = (float*)(ws + o_uf);
  float* x1  = (float*)(ws + o_x1);
  bf16* catb = (bf16*)(ws + o_cat);
  bf16* k1   = (bf16*)(ws + o_k1);
  bf16* xf   = (bf16*)(ws + o_xf);
  float* y   = (float*)(ws + o_y);
  bf16* ebuf = (bf16*)(ws + o_e);
  bf16* dbuf = (bf16*)(ws + o_d);
  bf16* pbuf = (bf16*)(ws + o_p);

  const long SS64 = 64L * HW, SS96 = 96L * HW, SS128 = 128L * HW, SS160 = 160L * HW,
             SS16 = 16L * HW, SS32 = 32L * HW;

  // 0) pack GEMM weights to bf16
  pack_weights_kernel<<<1, 256, 0, stream>>>(c11aw, c11abv, c1w, c1b, c211w, c211b,
                                             c2cw, kba1w, c3w, few, fpw,
                                             W1, b1, Wc2c, Wkb1, Wc3, Wfe, Wfp);
  // 1) n1 rms + channel means -> SCA gate
  sumsq_kernel<float><<<512, 256, 0, stream>>>(inp, SS64, 64, rssq, rsum);
  finalize1_kernel<<<8, 64, 0, stream>>>(rssq, rsum, n1w, n1b, scaw, scab, rms1, sca);
  // 2) fused norm + concat GEMM: xn -> [c11a | c1 | c211]  (WMMA)
  gemm1x1_kernel<64, true, false, false, true, false, false, false>
      <<<4096, 256, 0, stream>>>(
      inp, 64, SS64, n1w, n1b, rms1, W1, b1, 160,
      g1, SS160, nullptr, nullptr, 0, nullptr, 0);
  // 3) gated grouped 3x3 on normed input -> SimpleGate
  gate_kernel<<<dim3(256, 1, 8), 256, 0, stream>>>(inp, n1w, n1b, rms1, c2aw, c2ab, sg);
  // 4) x1 = grouped5x5(t1)
  grouped5x5_kernel<<<dim3(256, 16, 8), 256, 0, stream>>>(g1, SS160, c11bw, c11bb, x1, SS64);
  // 5) uf = dw3x3(t2); also write bf16 copy into catbuf ch 32..95
  dw3x3_kernel<false><<<dim3(256, 64, 8), 256, 0, stream>>>(
      g1 + 64L * HW, SS160, nullptr, nullptr, nullptr, c21w, c21b,
      uf, SS64, catb + SS32, SS96);
  // 6) att = attgamma*(c2c(sg)+b) + att0  -> catbuf ch 0..31  (WMMA, K padded 16->32)
  gemm1x1_kernel<32, false, false, false, false, true, true, false>
      <<<4096, 256, 0, stream>>>(
      sg, 16, SS16, nullptr, nullptr, nullptr, Wc2c, c2cb, 32,
      catb, SS96, attg, g1 + 128L * HW, SS160, nullptr, 0);
  // 7) k1 = kba1(cat(att,uf))  (WMMA, K=96)
  gemm1x1_kernel<96, false, false, false, false, false, false, false>
      <<<4096, 256, 0, stream>>>(
      catb, 96, SS96, nullptr, nullptr, nullptr, Wkb1, kba1b, 64,
      k1, SS64, nullptr, nullptr, 0, nullptr, 0);
  // 8) k = grouped5x5(k1); xf = (k*ga1 + uf)*x1*sca
  kba2_combine_kernel<<<dim3(256, 16, 8), 256, 0, stream>>>(
      k1, SS64, kba2w, kba2b, ga1, uf, x1, sca, xf);
  // 9) y = inp + beta*(c3(xf)+b)  (WMMA, fused residual)
  gemm1x1_kernel<64, false, false, true, false, true, false, true>
      <<<4096, 256, 0, stream>>>(
      xf, 64, SS64, nullptr, nullptr, nullptr, Wc3, c3b, 64,
      y, SS64, beta, nullptr, 0, inp, SS64);
  // 10) n2 rms over y
  sumsq_kernel<float><<<512, 256, 0, stream>>>(y, SS64, 64, rssq, nullptr);
  finalize_rms_kernel<<<8, 32, 0, stream>>>(rssq, 64, rms2);
  // 11) e = fe(norm2(y))  (WMMA, fused input affine)
  gemm1x1_kernel<64, true, false, false, true, false, false, false>
      <<<4096, 256, 0, stream>>>(
      y, 64, SS64, n2w, n2b, rms2, Wfe, feb, 128,
      ebuf, SS128, nullptr, nullptr, 0, nullptr, 0);
  // 12) fn1 rms over e
  sumsq_kernel<bf16><<<1024, 256, 0, stream>>>(ebuf, SS128, 128, rssq, nullptr);
  finalize_rms_kernel<<<8, 32, 0, stream>>>(rssq, 128, rmsE);
  // 13) d = dw3x3(gelu(norm(e)))
  dw3x3_kernel<true><<<dim3(256, 128, 8), 256, 0, stream>>>(
      ebuf, SS128, fn1w, fn1b, rmsE, fdww, fdwb, nullptr, 0, dbuf, SS128);
  // 14) fn2 rms over d
  sumsq_kernel<bf16><<<1024, 256, 0, stream>>>(dbuf, SS128, 128, rssq, nullptr);
  finalize_rms_kernel<<<8, 32, 0, stream>>>(rssq, 128, rmsD);
  // 15) p = fp(gelu(norm(d)))  (WMMA, fused affine+GELU staging, K=128)
  gemm1x1_kernel<128, false, true, false, true, false, false, false>
      <<<4096, 256, 0, stream>>>(
      dbuf, 128, SS128, fn2w, fn2b, rmsD, Wfp, fpb, 64,
      pbuf, SS64, nullptr, nullptr, 0, nullptr, 0);
  // 16) fn3 rms over p
  sumsq_kernel<bf16><<<512, 256, 0, stream>>>(pbuf, SS64, 64, rssq, nullptr);
  finalize_rms_kernel<<<8, 32, 0, stream>>>(rssq, 64, rmsP);
  // 17) out = y + (norm2(y) + norm3(p)*fls) * gamma
  final_kernel<<<131072, 256, 0, stream>>>(y, pbuf, n2w, n2b, rms2,
                                           fn3w, fn3b, rmsP, fls, gamma,
                                           (float*)d_out);
}
